// LayerNodeAttention_weight_30142080483328
// MI455X (gfx1250) — matrix operand order repaired and checked by reading, not compiled, not run on hardware
//
#include <hip/hip_runtime.h>
#include <hip/hip_bf16.h>

// ---------------------------------------------------------------------------
// GAT-style layer-node attention, fused for MI455X (gfx1250, wave32, WMMA).
//
// Key observation: argsort(nodes_ori) is the inverse permutation of nodes_ori
// and the final X[nodes_ori] undoes it, so out[i] depends only on
// node_features[:, i, :].  No gather needed.
//
// Compute-bound GEMM (26.2 GFLOP vs ~256MB traffic -> ~100 FLOP/B).  We use
// split-precision FP32 = bf16_hi + bf16_lo and 3x v_wmma_f32_16x16x32_bf16
// per K=32 step (hi*hi + lo*hi + hi*lo), ~2.7x the math rate of the native
// K=4 fp32 WMMA with ~2^-16 relative error (harmless after tanh + normalized
// convex combination).
// ---------------------------------------------------------------------------

typedef __attribute__((ext_vector_type(16))) __bf16 v16bf;
typedef __attribute__((ext_vector_type(8)))  __bf16 v8bf;
typedef __attribute__((ext_vector_type(4)))  __bf16 v4bf;
typedef __attribute__((ext_vector_type(8)))  float  v8f;

#define N_NODES   50000
#define M_LAYERS  4
#define FDIM      256
#define ALPHA     0.2f

#define NODES_PER_BLK 16
#define ROWS          64            // M_LAYERS * NODES_PER_BLK
#define A_STRIDE      264           // bf16 elems/row (+8 pad: 528B row -> 4-bank skew, conflict-free b128)
#define H_STRIDE      260           // f32 elems/row  (+4 pad)

// dynamic-LDS layout (union: A region dead before H region is written)
#define A_HI_OFF   0
#define A_LO_OFF   (ROWS * A_STRIDE * 2)                 // 33792
#define A_END      (2 * ROWS * A_STRIDE * 2)             // 67584
#define H_OFF      0
#define PART_OFF   (ROWS * H_STRIDE * 4)                 // 66560
#define WN_OFF     (PART_OFF + 5 * 16 * 16 * 4)          // 71680
#define SMEM_BYTES 73728

// --- prep: Wt[g][f] = trans[f][g], split into bf16 hi/lo (256KB, L2-resident)
__global__ __launch_bounds__(256) void prep_wt_kernel(
    const float* __restrict__ w, __bf16* __restrict__ hi, __bf16* __restrict__ lo) {
  int idx = blockIdx.x * 256 + threadIdx.x;   // 0..65535
  int g = idx >> 8, f = idx & 255;
  float v = w[f * FDIM + g];
  __bf16 h = (__bf16)v;
  hi[g * FDIM + f] = h;
  lo[g * FDIM + f] = (__bf16)(v - (float)h);
}

__global__ __launch_bounds__(256) void gat_fused_kernel(
    const float* __restrict__ nf,        // (M, N, F)
    const float* __restrict__ att,       // (2F,)
    const float* __restrict__ bias,      // (F,)
    const __bf16* __restrict__ wt_hi,    // (F=g, F=f) transposed trans, hi part
    const __bf16* __restrict__ wt_lo,
    float* __restrict__ out)             // (N, F)
{
  extern __shared__ __attribute__((aligned(16))) char smem[];
  __bf16* Ahi  = (__bf16*)(smem + A_HI_OFF);
  __bf16* Alo  = (__bf16*)(smem + A_LO_OFF);
  float*  Hs   = (float*)(smem + H_OFF);
  float*  part = (float*)(smem + PART_OFF);   // part[5][16][16]
  float*  wn   = (float*)(smem + WN_OFF);     // wn[4][16]

  const int tid   = threadIdx.x;
  const int lane  = tid & 31;
  const int wv    = tid >> 5;           // wave id 0..7 (uniform)
  const int nloc  = lane & 15;
  const int khalf = lane >> 4;          // 0: lanes 0-15, 1: lanes 16-31
  const int n0    = blockIdx.x * NODES_PER_BLK;

  // -------- Phase 0: load 64x256 fp32 A-block, split to bf16 hi/lo in LDS --
  // row = m*16 + node_local  (so row-tile index == layer m)
#pragma unroll
  for (int i = 0; i < 16; ++i) {
    int idx = tid + i * 256;            // 0..4095 float4 slots
    int row = idx >> 6;                 // 0..63
    int c4  = idx & 63;                 // float4 within row
    int m = row >> 4, nl = row & 15;
    const float4 x =
        *(const float4*)(nf + ((size_t)m * N_NODES + (n0 + nl)) * FDIM + c4 * 4);
    __bf16 h0 = (__bf16)x.x, h1 = (__bf16)x.y, h2 = (__bf16)x.z, h3 = (__bf16)x.w;
    v4bf hv = {h0, h1, h2, h3};
    v4bf lv = {(__bf16)(x.x - (float)h0), (__bf16)(x.y - (float)h1),
               (__bf16)(x.z - (float)h2), (__bf16)(x.w - (float)h3)};
    *(v4bf*)(Ahi + row * A_STRIDE + c4 * 4) = hv;
    *(v4bf*)(Alo + row * A_STRIDE + c4 * 4) = lv;
  }
  __syncthreads();

  // -------- Phase 1: WMMA K-loop.  Wave w owns col-tiles {2w,2w+1} x row-tiles 0..3
  v8f acc[4][2];
#pragma unroll
  for (int r = 0; r < 4; ++r)
#pragma unroll
    for (int cc = 0; cc < 2; ++cc)
      acc[r][cc] = (v8f){0.f, 0.f, 0.f, 0.f, 0.f, 0.f, 0.f, 0.f};

  const int c0 = wv * 2;
#pragma unroll
  for (int ks = 0; ks < 8; ++ks) {
    const int kb = ks * 32;
    // B fragments: lane covers column N=lane%16, K = kb + khalf*16 .. +15,
    // contiguous in Wt row-major (g-major) -> two b128 global loads each.
    v16bf Bhi[2], Blo[2];
#pragma unroll
    for (int cc = 0; cc < 2; ++cc) {
      const int g = (c0 + cc) * 16 + nloc;
      Bhi[cc] = *(const v16bf*)(wt_hi + (size_t)g * FDIM + kb + khalf * 16);
      Blo[cc] = *(const v16bf*)(wt_lo + (size_t)g * FDIM + kb + khalf * 16);
    }
#pragma unroll
    for (int r = 0; r < 4; ++r) {
      // A fragment (contiguous K order, matching B): lanes 0-15 hold K {0..15},
      // lanes 16-31 hold K {16..31} of row M=lane%16 -> two b128 LDS loads.
      const int arow = r * 16 + nloc;
      const int off  = khalf * 16;
      union { v16bf v; v8bf h[2]; } ah, al;
      ah.h[0] = *(const v8bf*)(Ahi + arow * A_STRIDE + kb + off);
      ah.h[1] = *(const v8bf*)(Ahi + arow * A_STRIDE + kb + off + 8);
      al.h[0] = *(const v8bf*)(Alo + arow * A_STRIDE + kb + off);
      al.h[1] = *(const v8bf*)(Alo + arow * A_STRIDE + kb + off + 8);
#pragma unroll
      for (int cc = 0; cc < 2; ++cc) {
        acc[r][cc] = __builtin_amdgcn_wmma_f32_16x16x32_bf16(
            false, ah.v, false, Bhi[cc], (short)0, acc[r][cc], false, false);
        acc[r][cc] = __builtin_amdgcn_wmma_f32_16x16x32_bf16(
            false, al.v, false, Bhi[cc], (short)0, acc[r][cc], false, false);
        acc[r][cc] = __builtin_amdgcn_wmma_f32_16x16x32_bf16(
            false, ah.v, false, Blo[cc], (short)0, acc[r][cc], false, false);
      }
    }
  }
  __syncthreads();   // A region dead everywhere; safe to overwrite with H

  // -------- Phase 2: bias + tanh, spill H to LDS (C layout: VGPR v -> M=v+8*khalf)
#pragma unroll
  for (int cc = 0; cc < 2; ++cc) {
    const int gc = (c0 + cc) * 16;
    const float bv = bias[gc + nloc];
#pragma unroll
    for (int r = 0; r < 4; ++r)
#pragma unroll
      for (int v = 0; v < 8; ++v) {
        const int node = v + khalf * 8;
        Hs[(r * 16 + node) * H_STRIDE + gc + nloc] = tanhf(acc[r][cc][v] + bv);
      }
  }
  __syncthreads();

  // -------- Phase 3a: attention dot partials. thread -> (node=tid>>4, seg=tid&15)
  {
    const int node = tid >> 4, seg = tid & 15, f0 = seg * 16;
    float p0 = 0.f, p1 = 0.f, p2 = 0.f, p3 = 0.f, p4 = 0.f;
#pragma unroll
    for (int j = 0; j < 16; ++j) {
      const float w1 = att[f0 + j];
      const float w2 = att[FDIM + f0 + j];
      const float h0 = Hs[(0 * 16 + node) * H_STRIDE + f0 + j];
      p0 += h0 * w1;
      p1 += h0 * w2;
      p2 += Hs[(1 * 16 + node) * H_STRIDE + f0 + j] * w2;
      p3 += Hs[(2 * 16 + node) * H_STRIDE + f0 + j] * w2;
      p4 += Hs[(3 * 16 + node) * H_STRIDE + f0 + j] * w2;
    }
    part[(0 * 16 + node) * 16 + seg] = p0;
    part[(1 * 16 + node) * 16 + seg] = p1;
    part[(2 * 16 + node) * 16 + seg] = p2;
    part[(3 * 16 + node) * 16 + seg] = p3;
    part[(4 * 16 + node) * 16 + seg] = p4;
  }
  __syncthreads();

  // -------- Phase 3b: per-node weights (16 threads; tiny)
  if (tid < 16) {
    const int nd = tid;
    float s0 = 0.f, tm0 = 0.f, tm1 = 0.f, tm2 = 0.f, tm3 = 0.f;
#pragma unroll
    for (int sg = 0; sg < 16; ++sg) {
      s0  += part[(0 * 16 + nd) * 16 + sg];
      tm0 += part[(1 * 16 + nd) * 16 + sg];
      tm1 += part[(2 * 16 + nd) * 16 + sg];
      tm2 += part[(3 * 16 + nd) * 16 + sg];
      tm3 += part[(4 * 16 + nd) * 16 + sg];
    }
    float tm[4] = {tm0, tm1, tm2, tm3};
    float wsum = 0.f, wvv[4];
#pragma unroll
    for (int m = 0; m < 4; ++m) {
      const float s  = s0 + tm[m];
      const float lr = s > 0.f ? s : ALPHA * s;
      float w = fminf(__expf(lr), 1.0f);
      if (m == 0) w = 0.f;                 // LP row zeroed
      wvv[m] = w;
      wsum  += w;
    }
#pragma unroll
    for (int m = 0; m < 4; ++m) wn[m * 16 + nd] = wvv[m] / wsum;
  }
  __syncthreads();

  // -------- Phase 3c: out[n] = h0 + sum_m wn[m]*h[m]  (wn[0]==0), coalesced f4 stores
  {
    const int nd = tid >> 4, sg = tid & 15, fb = sg * 16;
    const float w1 = wn[1 * 16 + nd], w2 = wn[2 * 16 + nd], w3 = wn[3 * 16 + nd];
    float* orow = out + (size_t)(n0 + nd) * FDIM;
#pragma unroll
    for (int j = 0; j < 16; j += 4) {
      const float4 h0 = *(const float4*)&Hs[(0 * 16 + nd) * H_STRIDE + fb + j];
      const float4 h1 = *(const float4*)&Hs[(1 * 16 + nd) * H_STRIDE + fb + j];
      const float4 h2 = *(const float4*)&Hs[(2 * 16 + nd) * H_STRIDE + fb + j];
      const float4 h3 = *(const float4*)&Hs[(3 * 16 + nd) * H_STRIDE + fb + j];
      float4 o;
      o.x = h0.x + w1 * h1.x + w2 * h2.x + w3 * h3.x;
      o.y = h0.y + w1 * h1.y + w2 * h2.y + w3 * h3.y;
      o.z = h0.z + w1 * h1.z + w2 * h2.z + w3 * h3.z;
      o.w = h0.w + w1 * h1.w + w2 * h2.w + w3 * h3.w;
      *(float4*)(orow + fb + j) = o;
    }
  }
}

extern "C" void kernel_launch(void* const* d_in, const int* in_sizes, int n_in,
                              void* d_out, int out_size, void* d_ws, size_t ws_size,
                              hipStream_t stream) {
  const float* nf    = (const float*)d_in[0];
  // d_in[1] (nodes_ori) is provably unused: the two permutations cancel.
  const float* trans = (const float*)d_in[2];
  const float* att   = (const float*)d_in[3];
  const float* bias  = (const float*)d_in[4];
  float* out = (float*)d_out;

  __bf16* wt_hi = (__bf16*)d_ws;             // 256*256 bf16 = 128KB
  __bf16* wt_lo = wt_hi + FDIM * FDIM;       // +128KB (workspace >= 256KB assumed)

  prep_wt_kernel<<<FDIM * FDIM / 256, 256, 0, stream>>>(trans, wt_hi, wt_lo);

  const int nblocks = N_NODES / NODES_PER_BLK;   // 50000/16 = 3125 exact
  gat_fused_kernel<<<nblocks, 256, SMEM_BYTES, stream>>>(nf, att, bias, wt_hi,
                                                         wt_lo, out);
}
